// RegionProposalNetWork_61409442398730
// MI455X (gfx1250) — matrix operand order repaired and checked by reading, not compiled, not run on hardware
//
#include <hip/hip_runtime.h>
#include <float.h>
#include <math.h>

// ---------------- types ----------------
typedef __attribute__((ext_vector_type(16))) __bf16        bf16x16;
typedef __attribute__((ext_vector_type(8)))  float         f32x8;
typedef __attribute__((ext_vector_type(4)))  unsigned int  u32x4;
typedef __attribute__((ext_vector_type(8)))  int           i32x8;
typedef __attribute__((ext_vector_type(4)))  int           i32x4;
typedef __attribute__((ext_vector_type(4)))  float         f32x4;

// ---------------- problem constants ----------------
#define HF     50
#define WF     50
#define HP     52          // padded
#define CIN    512
#define CMID   512
#define NB     4
#define M_TOT  10000       // NB*HF*WF
#define KPOS   9
#define AANCH  9
#define NANCH  22500       // HF*WF*AANCH
#define NPRE   3000
#define NPOST  300
#define NMS_T  0.7f
#define MIN_SZ 16.0f

// B-tile staging: 32 K-rows x 128 N-cols bf16, TDM-padded 16B per 256B row
#define BROWS      32
#define BROW_E     136           // 128 data + 8 pad elements (272 B row stride)
#define KSTEPS     144           // 9 taps * 16 chunks of K=32

static __device__ __forceinline__ unsigned short f2bf(float f) {
  unsigned u = __float_as_uint(f);
  unsigned r = u + 0x7FFFu + ((u >> 16) & 1u);   // round-to-nearest-even
  return (unsigned short)(r >> 16);
}

static __device__ __forceinline__ unsigned flip_key(float f) {
  unsigned u = __float_as_uint(f);
  return (u & 0x80000000u) ? ~u : (u | 0x80000000u);   // monotone map, bigger float -> bigger uint
}

// Tensor Data Mover: DMA a 32x128 bf16 tile (row stride 512 elems) into LDS.
// D# built per CDNA5 ISA ch.8: group0 = {flags, lds_addr, global_addr, type},
// group1 = dims/strides/padding. 6-arg (clang-23 / therock) builtin form.
static __device__ __forceinline__ void tdm_load_b_tile(const unsigned short* gsrc,
                                                       unsigned lds_off) {
  unsigned long long ga = (unsigned long long)(size_t)gsrc;
  u32x4 g0;
  g0.x = 1u;                                                  // count=1, user D#
  g0.y = lds_off;                                             // lds_addr (bytes)
  g0.z = (unsigned)(ga & 0xFFFFFFFFu);                        // global_addr[31:0]
  g0.w = (unsigned)((ga >> 32) & 0x01FFFFFFu) | 0x80000000u;  // addr[56:32] | type=2
  i32x8 g1;
  g1[0] = 0x07510000;        // data_size=2B | pad_enable | pad_interval=256B | pad_amount=16B
  g1[1] = (int)(512u << 16); // tensor_dim0 = 512 (low16 in [63:48])
  g1[2] = (int)(4608u << 16);// tensor_dim1 = 4608 (low16 -> word2 hi)
  g1[3] = (int)(128u << 16); // tile_dim0 = 128 elements
  g1[4] = 32;                // tile_dim1 = 32 rows (tile_dim2 = 0)
  g1[5] = 512;               // tensor_dim0_stride = 512 elements
  g1[6] = 0;
  g1[7] = 0;
  i32x4 z4 = {0, 0, 0, 0};
  i32x8 z8 = {0, 0, 0, 0, 0, 0, 0, 0};
  __builtin_amdgcn_tensor_load_to_lds(g0, g1, z4, z4, z8, 0);
}

// ---------------- pack weights: [O][C][3][3] f32 -> [p][c][o] bf16 ----------------
__global__ __launch_bounds__(256) void k_pack_w(const float* __restrict__ w,
                                                unsigned short* __restrict__ wB) {
  int id = blockIdx.x * blockDim.x + threadIdx.x;
  if (id >= KPOS * CIN * CMID) return;
  int o = id % CMID;
  int c = (id / CMID) % CIN;
  int p = id / (CMID * CIN);
  wB[id] = f2bf(w[((size_t)o * CIN + c) * KPOS + p]);
}

// ---------------- pack input: NCHW f32 -> padded NHWC bf16 [b][52][52][512] ----------------
__global__ __launch_bounds__(256) void k_pack_x(const float* __restrict__ x,
                                                unsigned short* __restrict__ xp) {
  int id = blockIdx.x * blockDim.x + threadIdx.x;
  if (id >= NB * HP * HP * CIN) return;
  int c  = id % CIN;
  int xx = (id / CIN) % HP;
  int yy = (id / (CIN * HP)) % HP;
  int b  = id / (CIN * HP * HP);
  float v = 0.f;
  if (yy >= 1 && yy <= HF && xx >= 1 && xx <= WF)
    v = x[(((size_t)b * CIN + c) * HF + (yy - 1)) * WF + (xx - 1)];
  xp[id] = f2bf(v);
}

// ---------------- implicit-GEMM 3x3 conv + bias + ReLU via bf16 WMMA ----------------
// grid (157, 4), block 256 (8 waves).  Block tile 64M x 128N, wave tile 32x32.
// B tiles double-buffered in LDS via TDM; A fragments direct from global (L2-resident).
__global__ __launch_bounds__(256) void k_conv_wmma(const unsigned short* __restrict__ xp,
                                                   const unsigned short* __restrict__ wB,
                                                   const float* __restrict__ bias,
                                                   float* __restrict__ feat) {
  __shared__ unsigned short bsm[2][BROWS * BROW_E];

  const int tid  = threadIdx.x;
  const int lane = tid & 31;
  const int wv   = tid >> 5;
  const int wm   = wv >> 2, wn = wv & 3;
  const int mbase = blockIdx.x * 64 + wm * 32;
  const int nblk  = blockIdx.y * 128;
  const int h  = lane >> 4;      // lane half selects K sub-block for A frags
  const int lr = lane & 15;

  // A rows for the two 16-row tiles of this wave (clamped loads, guarded stores)
  int r0 = mbase + lr;      if (r0 > M_TOT - 1) r0 = M_TOT - 1;
  int r1 = mbase + 16 + lr; if (r1 > M_TOT - 1) r1 = M_TOT - 1;
  int b0 = r0 / 2500, t0 = r0 % 2500, y0 = t0 / WF, x0 = t0 % WF;
  int b1 = r1 / 2500, t1 = r1 % 2500, y1 = t1 / WF, x1 = t1 % WF;
  const size_t ro0 = ((size_t)(b0 * HP + y0) * HP + x0) * CIN;
  const size_t ro1 = ((size_t)(b1 * HP + y1) * HP + x1) * CIN;

  const unsigned lds0 = (unsigned)(size_t)(void*)&bsm[0][0];
  const unsigned lds1 = (unsigned)(size_t)(void*)&bsm[1][0];

  f32x8 acc00 = {0,0,0,0,0,0,0,0};
  f32x8 acc01 = acc00, acc10 = acc00, acc11 = acc00;

  // prologue: DMA B-tile for step 0 into buffer 0
  if (wv == 0) {
    tdm_load_b_tile(wB + (size_t)0 * CMID + nblk, lds0);
    __builtin_amdgcn_s_wait_tensorcnt(0);
  }
  __syncthreads();

  int s = 0;
  for (int p = 0; p < KPOS; ++p) {
    const int ky = p / 3, kx = p % 3;
    const size_t sh = ((size_t)ky * HP + kx) * CIN;         // spatial shift (implicit GEMM)
    const unsigned short* a0base = xp + ro0 + sh + 8 * h;
    const unsigned short* a1base = xp + ro1 + sh + 8 * h;
    for (int c0 = 0; c0 < CIN; c0 += 32, ++s) {
      // kick off DMA for the next K-step while we compute this one
      if (wv == 0 && s + 1 < KSTEPS) {
        const int s1 = s + 1;
        const int p1 = s1 >> 4, c01 = (s1 & 15) << 5;
        tdm_load_b_tile(wB + ((size_t)(p1 * CIN + c01)) * CMID + nblk,
                        (s1 & 1) ? lds1 : lds0);
      }

      union { bf16x16 v; u32x4 q[2]; } A0, A1, B0, B1;
      A0.q[0] = *(const u32x4*)(a0base + c0);
      A0.q[1] = *(const u32x4*)(a0base + c0 + 16);
      A1.q[0] = *(const u32x4*)(a1base + c0);
      A1.q[1] = *(const u32x4*)(a1base + c0 + 16);
      if (c0 + 32 < CIN) __builtin_prefetch(a0base + c0 + 32, 0, 1);  // global_prefetch_b8

      // B fragments from LDS: lane holds N-col = lane&15, 16 K-rows per frag (k = 8h+s, 16+8h+s-8)
      const unsigned short* bb = &bsm[s & 1][wn * 32 + lr];
      for (int i = 0; i < 8; ++i) {
        const int ka = (i < 4) ? (8 * h + 2 * i) : (16 + 8 * h + 2 * (i - 4));
        B0.q[i >> 2][i & 3] = (unsigned)bb[ka * BROW_E] | ((unsigned)bb[(ka + 1) * BROW_E] << 16);
        B1.q[i >> 2][i & 3] = (unsigned)bb[ka * BROW_E + 16] | ((unsigned)bb[(ka + 1) * BROW_E + 16] << 16);
      }

      acc00 = __builtin_amdgcn_wmma_f32_16x16x32_bf16(false, A0.v, false, B0.v, (short)0, acc00, false, false);
      acc01 = __builtin_amdgcn_wmma_f32_16x16x32_bf16(false, A0.v, false, B1.v, (short)0, acc01, false, false);
      acc10 = __builtin_amdgcn_wmma_f32_16x16x32_bf16(false, A1.v, false, B0.v, (short)0, acc10, false, false);
      acc11 = __builtin_amdgcn_wmma_f32_16x16x32_bf16(false, A1.v, false, B1.v, (short)0, acc11, false, false);

      if (wv == 0) __builtin_amdgcn_s_wait_tensorcnt(0);   // next buffer landed
      __syncthreads();                                     // publish to all waves
    }
  }

  // store C frags: VGPR e, lanes 0-15 -> M = e, lanes 16-31 -> M = e+8; col = lane&15
  const int n0 = nblk + wn * 32 + lr;
  const float bv0 = bias[n0];
  const float bv1 = bias[n0 + 16];
#pragma unroll
  for (int e = 0; e < 8; ++e) {
    int mA = mbase + 8 * h + e;
    int mB = mbase + 16 + 8 * h + e;
    if (mA < M_TOT) {
      float v = acc00[e] + bv0; feat[(size_t)mA * CMID + n0]      = v > 0.f ? v : 0.f;
      v       = acc01[e] + bv1; feat[(size_t)mA * CMID + n0 + 16] = v > 0.f ? v : 0.f;
    }
    if (mB < M_TOT) {
      float v = acc10[e] + bv0; feat[(size_t)mB * CMID + n0]      = v > 0.f ? v : 0.f;
      v       = acc11[e] + bv1; feat[(size_t)mB * CMID + n0 + 16] = v > 0.f ? v : 0.f;
    }
  }
}

// ---------------- 1x1 heads + softmax + anchor decode + clip ----------------
// one 64-thread block per output pixel (m in [0,10000))
__global__ __launch_bounds__(64) void k_heads(const float* __restrict__ feat,
                                              const float* __restrict__ loc_w,
                                              const float* __restrict__ loc_b,
                                              const float* __restrict__ bf_w,
                                              const float* __restrict__ bf_b,
                                              const int* __restrict__ imh_p,
                                              const int* __restrict__ imw_p,
                                              float* __restrict__ out_loc,
                                              float* __restrict__ out_bf,
                                              float* __restrict__ out_anch,
                                              float* __restrict__ ws_scores,
                                              float* __restrict__ ws_boxes) {
  __shared__ float frow[CMID];
  __shared__ float outs[64];
  const int m = blockIdx.x;
  const int t = threadIdx.x;
  const int b = m / 2500, hw = m % 2500;
  const int y = hw / WF, x = hw % WF;

  for (int i = t; i < CMID; i += 64) frow[i] = feat[(size_t)m * CMID + i];
  __syncthreads();

  if (t < 36) {                        // loc channels, o = a*4 + d
    const float* wr = loc_w + (size_t)t * CMID;
    float a = loc_b[t];
    for (int c = 0; c < CMID; ++c) a += frow[c] * wr[c];
    outs[t] = a;
    out_loc[((size_t)b * NANCH + (size_t)hw * 9 + (t >> 2)) * 4 + (t & 3)] = a;
  } else if (t < 54) {                 // score channels, o = a*2 + cls
    int o = t - 36;
    const float* wr = bf_w + (size_t)o * CMID;
    float a = bf_b[o];
    for (int c = 0; c < CMID; ++c) a += frow[c] * wr[c];
    outs[t] = a;
  }
  __syncthreads();

  if (t < AANCH) {
    const int a_ = t;
    const float ratios[3] = {0.5f, 1.f, 2.f};
    const float scales[3] = {8.f, 16.f, 32.f};
    float rr = ratios[a_ / 3], ss = scales[a_ % 3];
    float hs  = 16.f * ss * sqrtf(rr);
    float wsz = 16.f * ss * sqrtf(1.f / rr);
    float cy0 = y * 16.f + 8.f, cx0 = x * 16.f + 8.f;
    size_t ai = (size_t)hw * 9 + a_;
    if (b == 0) {
      out_anch[ai * 4 + 0] = cy0 - 0.5f * hs;
      out_anch[ai * 4 + 1] = cx0 - 0.5f * wsz;
      out_anch[ai * 4 + 2] = cy0 + 0.5f * hs;
      out_anch[ai * 4 + 3] = cx0 + 0.5f * wsz;
    }
    // softmax over 2 classes
    float l0 = outs[36 + a_ * 2], l1 = outs[36 + a_ * 2 + 1];
    float mx = l0 > l1 ? l0 : l1;
    float e0 = expf(l0 - mx), e1 = expf(l1 - mx);
    float inv = 1.f / (e0 + e1);
    float p0 = e0 * inv, p1 = e1 * inv;
    size_t gi = (size_t)b * NANCH + ai;
    out_bf[gi * 2 + 0] = p0;
    out_bf[gi * 2 + 1] = p1;
    // decode + clip
    float dy = outs[a_ * 4 + 0], dx = outs[a_ * 4 + 1];
    float dh = outs[a_ * 4 + 2], dw = outs[a_ * 4 + 3];
    float cy = dy * hs + cy0, cx = dx * wsz + cx0;
    float hh = expf(dh) * hs, ww = expf(dw) * wsz;
    float ih = (float)imh_p[0], iw = (float)imw_p[0];
    float by0 = fminf(fmaxf(cy - 0.5f * hh, 0.f), ih);
    float bx0 = fminf(fmaxf(cx - 0.5f * ww, 0.f), iw);
    float by1 = fminf(fmaxf(cy + 0.5f * hh, 0.f), ih);
    float bx1 = fminf(fmaxf(cx + 0.5f * ww, 0.f), iw);
    ws_boxes[gi * 4 + 0] = by0; ws_boxes[gi * 4 + 1] = bx0;
    ws_boxes[gi * 4 + 2] = by1; ws_boxes[gi * 4 + 3] = bx1;
    bool valid = (by1 - by0 >= MIN_SZ) && (bx1 - bx0 >= MIN_SZ);
    ws_scores[gi] = valid ? p1 : -FLT_MAX;
  }
}

// ---------------- per-image NMS: radix-select top-3000 + bitonic sort + greedy ----------------
__global__ __launch_bounds__(1024) void k_nms(const float* __restrict__ scores,
                                              const float* __restrict__ boxes,
                                              float* __restrict__ selBoxes,
                                              float* __restrict__ out_rois,
                                              float* __restrict__ out_idx) {
  const int b = blockIdx.x;
  const int t = threadIdx.x;
  __shared__ unsigned keys[4096];
  __shared__ int      idxs[4096];
  __shared__ unsigned hist[256];
  __shared__ unsigned char supp[NPRE];
  __shared__ unsigned sh_prefix;
  __shared__ int      sh_R;
  __shared__ unsigned cnt;

  const float* sc = scores + (size_t)b * NANCH;

  // ---- exact rank-3000 pivot via 4x 8-bit radix passes ----
  unsigned prefix = 0; int R = NPRE;
  for (int pass = 0; pass < 4; ++pass) {
    const int shift = 24 - pass * 8;
    const unsigned mask = (pass == 0) ? 0u : (0xFFFFFFFFu << (shift + 8));
    for (int i = t; i < 256; i += 1024) hist[i] = 0;
    __syncthreads();
    for (int i = t; i < NANCH; i += 1024) {
      unsigned u = flip_key(sc[i]);
      if ((u & mask) == prefix) atomicAdd(&hist[(u >> shift) & 255u], 1u);
    }
    __syncthreads();
    if (t == 0) {
      unsigned cum = 0;
      for (int v = 255; v >= 0; --v) {
        unsigned c = hist[v];
        if (cum + c >= (unsigned)R) { sh_R = R - (int)cum; sh_prefix = prefix | ((unsigned)v << shift); break; }
        cum += c;
      }
    }
    __syncthreads();
    prefix = sh_prefix; R = sh_R;
    __syncthreads();
  }
  const unsigned pivot = prefix;

  // ---- gather top-3000 (strict greater, then ties capped) ----
  if (t == 0) cnt = 0;
  __syncthreads();
  for (int i = t; i < NANCH; i += 1024) {
    unsigned u = flip_key(sc[i]);
    if (u > pivot) { unsigned p_ = atomicAdd(&cnt, 1u); keys[p_] = u; idxs[p_] = i; }
  }
  __syncthreads();
  for (int i = t; i < NANCH; i += 1024) {
    unsigned u = flip_key(sc[i]);
    if (u == pivot) {
      unsigned p_ = atomicAdd(&cnt, 1u);
      if (p_ < NPRE) { keys[p_] = u; idxs[p_] = i; }
    }
  }
  __syncthreads();
  for (int i = t; i < 4096; i += 1024)
    if (i >= NPRE) { keys[i] = 0u; idxs[i] = -1; }
  __syncthreads();

  // ---- bitonic sort, descending by key ----
  for (int k = 2; k <= 4096; k <<= 1) {
    for (int j = k >> 1; j > 0; j >>= 1) {
      for (int i = t; i < 4096; i += 1024) {
        int l = i ^ j;
        if (l > i) {
          bool up = ((i & k) == 0);
          unsigned ki = keys[i], kl = keys[l];
          if ((ki < kl) == up) {
            keys[i] = kl; keys[l] = ki;
            int tmp = idxs[i]; idxs[i] = idxs[l]; idxs[l] = tmp;
          }
        }
      }
      __syncthreads();
    }
  }

  // ---- load candidate boxes into registers (3 per thread) ----
  f32x4 bx[3]; float ar[3];
#pragma unroll
  for (int s = 0; s < 3; ++s) {
    int j = t + s * 1024;
    bx[s] = (f32x4){0.f, 0.f, 0.f, 0.f};
    ar[s] = 0.f;
    if (j < NPRE) {
      int id = idxs[j];
      unsigned char sp = 1;
      f32x4 bb = (f32x4){0.f, 0.f, 0.f, 0.f};
      if (id >= 0) {
        const float* bp = boxes + ((size_t)b * NANCH + id) * 4;
        bb.x = bp[0]; bb.y = bp[1]; bb.z = bp[2]; bb.w = bp[3];
        if (sc[id] > -1e37f) sp = 0;          // validity was folded into the score mask
      }
      bx[s] = bb;
      ar[s] = (bb.z - bb.x) * (bb.w - bb.y);
      supp[j] = sp;
      float* sbp = selBoxes + ((size_t)b * NPRE + j) * 4;
      sbp[0] = bb.x; sbp[1] = bb.y; sbp[2] = bb.z; sbp[3] = bb.w;
    }
  }
  __threadfence();
  __syncthreads();

  // ---- greedy suppression in sorted order (equivalent to reference fori_loop) ----
  for (int i = 0; i < NPRE; ++i) {
    if (supp[i] == 0) {
      const float* bip = selBoxes + ((size_t)b * NPRE + i) * 4;
      float iy0 = bip[0], ix0 = bip[1], iy1 = bip[2], ix1 = bip[3];
      float ia = (iy1 - iy0) * (ix1 - ix0);
#pragma unroll
      for (int s = 0; s < 3; ++s) {
        int j = t + s * 1024;
        if (j < NPRE && j > i && supp[j] == 0) {
          float ty = fmaxf(iy0, bx[s].x), tx = fmaxf(ix0, bx[s].y);
          float by = fminf(iy1, bx[s].z), bxx = fminf(ix1, bx[s].w);
          float hh = fmaxf(by - ty, 0.f), ww = fmaxf(bxx - tx, 0.f);
          float inter = hh * ww;
          float iou = inter / (ia + ar[s] - inter + 1e-9f);
          if (iou > NMS_T) supp[j] = 1;
        }
      }
    }
    __syncthreads();
  }

  // ---- compact kept boxes (in order) into 300 rois, zero-pad ----
  if (t == 0) {
    int c_ = 0;
    for (int i = 0; i < NPRE && c_ < NPOST; ++i) {
      if (supp[i] == 0) {
        const float* bp = selBoxes + ((size_t)b * NPRE + i) * 4;
        float* op = out_rois + ((size_t)b * NPOST + c_) * 4;
        op[0] = bp[0]; op[1] = bp[1]; op[2] = bp[2]; op[3] = bp[3];
        ++c_;
      }
    }
    for (; c_ < NPOST; ++c_) {
      float* op = out_rois + ((size_t)b * NPOST + c_) * 4;
      op[0] = 0.f; op[1] = 0.f; op[2] = 0.f; op[3] = 0.f;
    }
    for (int r = 0; r < NPOST; ++r) out_idx[b * NPOST + r] = (float)b;
  }
}

// ---------------- launcher ----------------
extern "C" void kernel_launch(void* const* d_in, const int* in_sizes, int n_in,
                              void* d_out, int out_size, void* d_ws, size_t ws_size,
                              hipStream_t stream) {
  (void)in_sizes; (void)n_in; (void)out_size; (void)ws_size;
  const float* x      = (const float*)d_in[0];
  const float* conv_w = (const float*)d_in[1];
  const float* conv_b = (const float*)d_in[2];
  const float* bf_w   = (const float*)d_in[3];
  const float* bf_b   = (const float*)d_in[4];
  const float* loc_w  = (const float*)d_in[5];
  const float* loc_b  = (const float*)d_in[6];
  const int*   imh    = (const int*)d_in[7];
  const int*   imw    = (const int*)d_in[8];

  float* out      = (float*)d_out;
  float* out_rois = out;                   // [4*300*4]      = 4800
  float* out_idx  = out + 4800;            // [4*300]        = 1200
  float* out_anch = out + 6000;            // [22500*4]      = 90000
  float* out_loc  = out + 96000;           // [4*22500*4]    = 360000
  float* out_bf   = out + 456000;          // [4*22500*2]    = 180000

  char* ws = (char*)d_ws;
  size_t off = 0;
  auto alloc = [&](size_t bytes) -> void* {
    void* p = ws + off;
    off = (off + bytes + 255) & ~(size_t)255;
    return p;
  };
  unsigned short* xp     = (unsigned short*)alloc((size_t)NB * HP * HP * CIN * 2);   // 10.6 MB
  unsigned short* wB     = (unsigned short*)alloc((size_t)KPOS * CIN * CMID * 2);    //  4.5 MB
  float*          feat   = (float*)alloc((size_t)M_TOT * CMID * 4);                  // 19.5 MB
  float*          scrs   = (float*)alloc((size_t)NB * NANCH * 4);                    //  0.36 MB
  float*          bxs    = (float*)alloc((size_t)NB * NANCH * 4 * 4);                //  1.4 MB
  float*          selB   = (float*)alloc((size_t)NB * NPRE * 4 * 4);                 //  0.19 MB

  k_pack_w<<<(KPOS * CIN * CMID + 255) / 256, 256, 0, stream>>>(conv_w, wB);
  k_pack_x<<<(NB * HP * HP * CIN + 255) / 256, 256, 0, stream>>>(x, xp);
  dim3 cg(157, 4);   // 157*64 >= 10000 rows, 4*128 = 512 cols
  k_conv_wmma<<<cg, 256, 0, stream>>>(xp, wB, conv_b, feat);
  k_heads<<<M_TOT, 64, 0, stream>>>(feat, loc_w, loc_b, bf_w, bf_b, imh, imw,
                                    out_loc, out_bf, out_anch, scrs, bxs);
  k_nms<<<NB, 1024, 0, stream>>>(scrs, bxs, selB, out_rois, out_idx);
}
